// VGG_SNN_STDB_3341484556656
// MI455X (gfx1250) — matrix operand order repaired and checked by reading, not compiled, not run on hardware
//
#include <hip/hip_runtime.h>
#include <hip/hip_bf16.h>

// ---------------------------------------------------------------------------
// VGG-SNN forward on MI455X (gfx1250), bf16 WMMA with f32 accumulation.
// Spike activations are exact in bf16; weights converted f32->bf16 once per
// launch; conv(x,w0) is timestep-invariant and precomputed once.
// GEMMs: A operand is double-buffered through LDS with
// global_load_async_to_lds_b128 (ASYNCcnt) so the 4 N-waves of a WG share
// one copy of the A tile and DMA overlaps WMMA math. The LDS buffer address
// is passed into the DMA asm so alias analysis knows the asm writes it.
// ---------------------------------------------------------------------------

typedef __bf16 bf16;
typedef __attribute__((ext_vector_type(16))) __bf16 v16bf;
typedef __attribute__((ext_vector_type(8)))  float  v8f;

union FragAB { v16bf v; uint4 q[2]; };

// ------------------------------ GEMM (NT) ----------------------------------
// D[M,N] (f32 row-major, ldd) += A[M,K] (bf16 row-major, lda, K-major)
//                              x B[N,K] (bf16 row-major, ldb, K-major)^T
// 128 threads = 4 waves laid out 1(M) x 4(N). Wave tile 64M x 32N.
// Grid: (M/64, N/128). Requires M%64==0, N%128==0, K%64==0.
// A tile (64 x 64 K-chunk) staged in LDS via async DMA, double-buffered.
#define LDS_ROW   144                 // 64*2 bytes + 16B pad (bank-conflict-free)
#define LDS_BUF   (64 * LDS_ROW)      // 9216 B per buffer

__global__ __launch_bounds__(128)
void gemm_nt_bf16_acc(const bf16* __restrict__ A, const bf16* __restrict__ B,
                      float* __restrict__ D, int lda, int ldb, int ldd, int K)
{
    __shared__ char smem[2 * LDS_BUF];

    const int wave = threadIdx.x >> 5;   // N column of this wave in the WG
    const int lane = threadIdx.x & 31;
    const int r    = lane & 15;          // row (A) / col (B,D) within tile
    const int h    = lane >> 4;          // K-half selector

    const int m0 = blockIdx.x * 64;
    const int n0 = blockIdx.y * 128 + wave * 32;

    v8f acc[4][2];
#pragma unroll
    for (int mt = 0; mt < 4; ++mt)
#pragma unroll
        for (int nt = 0; nt < 2; ++nt)
#pragma unroll
            for (int v = 0; v < 8; ++v)
                acc[mt][nt][v] =
                    D[(size_t)(m0 + mt * 16 + h * 8 + v) * ldd + (n0 + nt * 16 + r)];

    // Async-DMA one 64x64 bf16 A chunk into LDS buffer `bufsel`.
    // 512 x 16B chunks; 128 threads x 4 issues. Row `row`, 16B chunk `c16`.
    // NOTE: smem's address is passed into the asm so the compiler knows the
    // DMA writes LDS (otherwise the ds reads get hoisted as never-written).
    auto stage = [&](int bufsel, int kb) {
#pragma unroll
        for (int j = 0; j < 4; ++j) {
            int q   = j * 128 + (int)threadIdx.x;
            int row = q >> 3;
            int c16 = q & 7;
            unsigned lofs = (unsigned)(bufsel * LDS_BUF + row * LDS_ROW + c16 * 16);
            unsigned long long g =
                (unsigned long long)(A + (size_t)(m0 + row) * lda + kb + c16 * 8);
            asm volatile("global_load_async_to_lds_b128 %0, %1, off"
                         :: "v"(lofs), "v"(g), "v"((void*)smem) : "memory");
        }
    };

    // B fragment per lane: row (n0+r) of the NxK weight, K = k + h*8 .. +7 and k + 16 + h*8 .. +7
    const bf16* bBase = B + (size_t)(n0 + r) * ldb + h * 8;
    // A fragment per lane (from LDS): row (mt*16 + r), bytes k*2 + h*16 (+32)
    const char* aLds = &smem[(size_t)r * LDS_ROW + h * 16];

    stage(0, 0);
    int buf = 0;
    for (int kb = 0; kb < K; kb += 64) {
        asm volatile("s_wait_asynccnt 0" ::: "memory");   // this wave's DMA done
        __syncthreads();                                  // publish LDS to all waves
        if (kb + 64 < K) stage(buf ^ 1, kb + 64);         // prefetch next chunk

        const char* aBuf = aLds + buf * LDS_BUF;
#pragma unroll
        for (int ks = 0; ks < 64; ks += 32) {
            FragAB af[4], bfr[2];
#pragma unroll
            for (int mt = 0; mt < 4; ++mt) {
                const char* p = aBuf + mt * 16 * LDS_ROW + ks * 2;
                af[mt].q[0] = *(const uint4*)(p);
                af[mt].q[1] = *(const uint4*)(p + 32);
            }
#pragma unroll
            for (int nt = 0; nt < 2; ++nt) {
                const bf16* p = bBase + (size_t)nt * 16 * ldb + kb + ks;
                bfr[nt].q[0] = *(const uint4*)(p);
                bfr[nt].q[1] = *(const uint4*)(p + 16);
            }
#pragma unroll
            for (int mt = 0; mt < 4; ++mt)
#pragma unroll
                for (int nt = 0; nt < 2; ++nt)
                    acc[mt][nt] = __builtin_amdgcn_wmma_f32_16x16x32_bf16(
                        false, af[mt].v, false, bfr[nt].v,
                        (short)0, acc[mt][nt], false, false);
        }
        __syncthreads();   // all waves done reading `buf` before it is refilled
        buf ^= 1;
    }

#pragma unroll
    for (int mt = 0; mt < 4; ++mt)
#pragma unroll
        for (int nt = 0; nt < 2; ++nt)
#pragma unroll
            for (int v = 0; v < 8; ++v)
                D[(size_t)(m0 + mt * 16 + h * 8 + v) * ldd + (n0 + nt * 16 + r)] =
                    acc[mt][nt][v];
}

// Final layer: mo(64x10) += s12(64x4096) x fc6r(16x4096, rows 10..15 zero)^T.
// One WG, 4 waves = one 16-row M tile each; masked store for n<10.
__global__ __launch_bounds__(128)
void gemm_fc6_acc(const bf16* __restrict__ A, const bf16* __restrict__ B,
                  float* __restrict__ D, int K)
{
    const int wave = threadIdx.x >> 5;   // M tile
    const int lane = threadIdx.x & 31;
    const int r = lane & 15, h = lane >> 4;
    const int m0 = wave * 16;

    v8f acc;
#pragma unroll
    for (int v = 0; v < 8; ++v)
        acc[v] = (r < 10) ? D[(m0 + h * 8 + v) * 10 + r] : 0.0f;

    const bf16* aBase = A + (size_t)(m0 + r) * K + h * 8;
    const bf16* bBase = B + (size_t)r * K + h * 8;

    for (int k = 0; k < K; k += 32) {
        FragAB af, bfr;
        af.q[0]  = *(const uint4*)(aBase + k);
        af.q[1]  = *(const uint4*)(aBase + k + 16);
        bfr.q[0] = *(const uint4*)(bBase + k);
        bfr.q[1] = *(const uint4*)(bBase + k + 16);
        acc = __builtin_amdgcn_wmma_f32_16x16x32_bf16(
            false, af.v, false, bfr.v, (short)0, acc, false, false);
    }
    if (r < 10) {
#pragma unroll
        for (int v = 0; v < 8; ++v)
            D[(m0 + h * 8 + v) * 10 + r] = acc[v];
    }
}

// --------------------------- elementwise kernels ---------------------------

__global__ void zero_f32(float* __restrict__ p, int n)
{
    int i = blockIdx.x * blockDim.x + threadIdx.x;
    if (i < n) p[i] = 0.0f;
}

// Layer 0: spike + 2x2 avg-pool + membrane update (m0 += C0, timestep-invariant
// conv already precomputed). m0/C0 are NHWC (64,32,32,64); h0 NHWC (64,16,16,64) bf16.
__global__ void spike_pool_m0(float* __restrict__ m0, const float* __restrict__ C0,
                              bf16* __restrict__ h0,
                              const float* __restrict__ th, const float* __restrict__ lk)
{
    int i = blockIdx.x * blockDim.x + threadIdx.x;   // 16384 pooled pixels * 64 ch
    if (i >= 16384 * 64) return;
    int c  = i & 63;
    int p  = i >> 6;
    int b  = p >> 8;
    int y8 = (p >> 4) & 15;
    int x8 = p & 15;
    float T = th[0], L = lk[0];
    float sum = 0.0f;
#pragma unroll
    for (int dy = 0; dy < 2; ++dy)
#pragma unroll
        for (int dx = 0; dx < 2; ++dx) {
            int yy = y8 * 2 + dy, xx = x8 * 2 + dx;
            size_t idx = ((size_t)((b * 32 + yy) * 32 + xx)) * 64 + c;
            float m = m0[idx];
            float s = (m > T) ? 1.0f : 0.0f;   // m/T - 1 > 0  (T > 0)
            m0[idx] = L * m - T * s + C0[idx];
            sum += s;
        }
    h0[(size_t)p * 64 + c] = (bf16)(sum * 0.25f);
}

// Generic spike + in-place leak/reset; writes bf16 spikes (fc layers and m3).
__global__ void spike_fc(float* __restrict__ m, bf16* __restrict__ s, int n,
                         const float* __restrict__ th, const float* __restrict__ lk,
                         int idx)
{
    int i = blockIdx.x * blockDim.x + threadIdx.x;
    if (i >= n) return;
    float T = th[idx], L = lk[idx];
    float v = m[i];
    float sp = (v > T) ? 1.0f : 0.0f;
    m[i] = L * v - T * sp;
    s[i] = (bf16)sp;
}

// Layer 6: spike + pool; m6 NHWC (64,16,16,128); h6 rows k' = (y8*8+x8)*128+c
// (fc0 weights were K-permuted to match this order).
__global__ void spike_pool_m6(float* __restrict__ m6, bf16* __restrict__ h6,
                              const float* __restrict__ th, const float* __restrict__ lk)
{
    int i = blockIdx.x * blockDim.x + threadIdx.x;   // 64*8*8*128
    if (i >= 524288) return;
    int c  = i & 127;
    int p  = i >> 7;
    int b  = p >> 6;
    int y8 = (p >> 3) & 7;
    int x8 = p & 7;
    float T = th[2], L = lk[2];
    float sum = 0.0f;
#pragma unroll
    for (int dy = 0; dy < 2; ++dy)
#pragma unroll
        for (int dx = 0; dx < 2; ++dx) {
            int yy = y8 * 2 + dy, xx = x8 * 2 + dx;
            size_t idx = ((size_t)((b * 16 + yy) * 16 + xx)) * 128 + c;
            float m = m6[idx];
            float s = (m > T) ? 1.0f : 0.0f;
            m6[idx] = L * m - T * s;
            sum += s;
        }
    h6[(size_t)b * 8192 + (size_t)(y8 * 8 + x8) * 128 + c] = (bf16)(sum * 0.25f);
}

// im2col for 3x3/s1/p1 conv on 16x16 NHWC bf16 maps. One thread copies one
// (ky,kx) channel chunk (C bf16 = C/8 x uint4). dst row = 9*C, K-order (ky,kx,ic).
__global__ void im2col16(const bf16* __restrict__ src, bf16* __restrict__ dst, int C)
{
    int i = blockIdx.x * blockDim.x + threadIdx.x;
    if (i >= 16384 * 9) return;
    int kk = i % 9;
    int p  = i / 9;
    int b = p >> 8, y = (p >> 4) & 15, x = p & 15;
    int ky = kk / 3, kx = kk % 3;
    int yy = y + ky - 1, xx = x + kx - 1;
    uint4* d = (uint4*)(dst + ((size_t)p * 9 + kk) * C);
    int nvec = C >> 3;
    if (yy >= 0 && yy < 16 && xx >= 0 && xx < 16) {
        const uint4* s = (const uint4*)(src + ((size_t)((b * 16 + yy) * 16 + xx)) * C);
        for (int j = 0; j < nvec; ++j) d[j] = s[j];
    } else {
        uint4 z; z.x = 0u; z.y = 0u; z.z = 0u; z.w = 0u;
        for (int j = 0; j < nvec; ++j) d[j] = z;
    }
}

// ----------------------- one-time precompute kernels -----------------------

// Direct 3->64 3x3 conv of the (timestep-invariant) input, NCHW in, NHWC out.
__global__ void conv0_direct(const float* __restrict__ x, const float* __restrict__ w0,
                             float* __restrict__ C0)
{
    int i = blockIdx.x * blockDim.x + threadIdx.x;   // 64*1024*64
    if (i >= 64 * 1024 * 64) return;
    int oc = i & 63;
    int p  = i >> 6;
    int b  = p >> 10;
    int y  = (p >> 5) & 31;
    int x0 = p & 31;
    float s = 0.0f;
    for (int ic = 0; ic < 3; ++ic) {
        const float* xp = x + (size_t)(b * 3 + ic) * 1024;
        const float* wp = w0 + (size_t)(oc * 3 + ic) * 9;
#pragma unroll
        for (int ky = 0; ky < 3; ++ky) {
            int yy = y + ky - 1;
            if (yy < 0 || yy > 31) continue;
#pragma unroll
            for (int kx = 0; kx < 3; ++kx) {
                int xx = x0 + kx - 1;
                if (xx < 0 || xx > 31) continue;
                s += xp[yy * 32 + xx] * wp[ky * 3 + kx];
            }
        }
    }
    C0[i] = s;
}

// fc0 f32->bf16 with K permuted from (c,y,x) flatten to our (y,x,c) NHWC flatten.
__global__ void cvt_fc0(const float* __restrict__ src, bf16* __restrict__ dst)
{
    int i = blockIdx.x * blockDim.x + threadIdx.x;   // 4096*8192
    if (i >= 4096 * 8192) return;
    int n  = i >> 13;
    int kp = i & 8191;
    int y = kp >> 10;
    int x = (kp >> 7) & 7;
    int c = kp & 127;
    int k = c * 64 + y * 8 + x;
    dst[i] = (bf16)src[(size_t)n * 8192 + k];
}

__global__ void cvt_plain(const float* __restrict__ src, bf16* __restrict__ dst, int n)
{
    int i = blockIdx.x * blockDim.x + threadIdx.x;
    if (i < n) dst[i] = (bf16)src[i];
}

// fc6 -> 16 rows (rows 10..15 zero-padded) for a full WMMA N tile.
__global__ void cvt_fc6(const float* __restrict__ src, bf16* __restrict__ dst)
{
    int i = blockIdx.x * blockDim.x + threadIdx.x;   // 16*4096
    if (i >= 16 * 4096) return;
    int n = i >> 12;
    dst[i] = (n < 10) ? (bf16)src[i] : (bf16)0.0f;
}

// conv weights (oc,ic,ky,kx) -> (oc,ky,kx,ic) bf16, matching im2col K order.
__global__ void cvt_w3(const float* __restrict__ src, bf16* __restrict__ dst)
{
    int i = blockIdx.x * blockDim.x + threadIdx.x;   // 128*576
    if (i >= 128 * 576) return;
    int oc = i / 576, t = i % 576;
    int ky = t / 192, kx = (t % 192) / 64, ic = t % 64;
    dst[i] = (bf16)src[(size_t)oc * 576 + ic * 9 + ky * 3 + kx];
}

__global__ void cvt_w6(const float* __restrict__ src, bf16* __restrict__ dst)
{
    int i = blockIdx.x * blockDim.x + threadIdx.x;   // 128*1152
    if (i >= 128 * 1152) return;
    int oc = i / 1152, t = i % 1152;
    int ky = t / 384, kx = (t % 384) / 128, ic = t % 128;
    dst[i] = (bf16)src[(size_t)oc * 1152 + ic * 9 + ky * 3 + kx];
}

// ------------------------------- launcher ----------------------------------

extern "C" void kernel_launch(void* const* d_in, const int* in_sizes, int n_in,
                              void* d_out, int out_size, void* d_ws, size_t ws_size,
                              hipStream_t stream)
{
    (void)in_sizes; (void)n_in; (void)out_size; (void)ws_size;

    const float* x   = (const float*)d_in[0];
    const float* w0  = (const float*)d_in[1];
    const float* w3  = (const float*)d_in[2];
    const float* w6  = (const float*)d_in[3];
    const float* fc0 = (const float*)d_in[4];
    const float* fc3 = (const float*)d_in[5];
    const float* fc6 = (const float*)d_in[6];
    const float* th  = (const float*)d_in[7];
    const float* lk  = (const float*)d_in[8];
    float* mo = (float*)d_out;   // (64,10) f32, accumulated over timesteps

    // --- workspace carve-up (all chunks 256B-aligned) ---
    char* base = (char*)d_ws;
    size_t off = 0;
    auto take = [&](size_t bytes) -> void* {
        void* p = base + off;
        off += (bytes + 255) & ~(size_t)255;
        return p;
    };
    bf16* fc0r = (bf16*)take((size_t)4096 * 8192 * 2);   // 67.1 MB, K-permuted
    bf16* fc3r = (bf16*)take((size_t)4096 * 4096 * 2);   // 33.6 MB
    bf16* fc6r = (bf16*)take((size_t)16 * 4096 * 2);     // padded to 16 rows
    bf16* w3r  = (bf16*)take((size_t)128 * 576 * 2);
    bf16* w6r  = (bf16*)take((size_t)128 * 1152 * 2);
    float* C0  = (float*)take((size_t)4194304 * 4);      // conv(x,w0), NHWC
    float* m0  = (float*)take((size_t)4194304 * 4);      // (64,32,32,64) NHWC
    float* m3  = (float*)take((size_t)2097152 * 4);      // (16384,128)
    float* m6  = (float*)take((size_t)2097152 * 4);
    float* m9  = (float*)take((size_t)262144 * 4);       // (64,4096)
    float* m12 = (float*)take((size_t)262144 * 4);
    bf16* h0   = (bf16*)take((size_t)1048576 * 2);       // pooled s0, (16384,64)
    bf16* s3   = (bf16*)take((size_t)2097152 * 2);       // (16384,128)
    bf16* h6   = (bf16*)take((size_t)524288 * 2);        // pooled s6, (64,8192)
    bf16* s9   = (bf16*)take((size_t)262144 * 2);
    bf16* s12  = (bf16*)take((size_t)262144 * 2);
    bf16* A3   = (bf16*)take((size_t)16384 * 576 * 2);   // im2col(h0)
    bf16* A6   = (bf16*)take((size_t)16384 * 1152 * 2);  // im2col(s3)

    // --- init state (ws/d_out are poisoned by the harness) ---
    zero_f32<<<16384, 256, 0, stream>>>(m0, 4194304);
    zero_f32<<<8192, 256, 0, stream>>>(m3, 2097152);
    zero_f32<<<8192, 256, 0, stream>>>(m6, 2097152);
    zero_f32<<<1024, 256, 0, stream>>>(m9, 262144);
    zero_f32<<<1024, 256, 0, stream>>>(m12, 262144);
    zero_f32<<<3, 256, 0, stream>>>(mo, 640);

    // --- one-time weight conversion + invariant conv ---
    cvt_fc0<<<131072, 256, 0, stream>>>(fc0, fc0r);
    cvt_plain<<<65536, 256, 0, stream>>>(fc3, fc3r, 16777216);
    cvt_fc6<<<256, 256, 0, stream>>>(fc6, fc6r);
    cvt_w3<<<288, 256, 0, stream>>>(w3, w3r);
    cvt_w6<<<576, 256, 0, stream>>>(w6, w6r);
    conv0_direct<<<16384, 256, 0, stream>>>(x, w0, C0);

    // --- 20 timesteps ---
    for (int t = 0; t < 20; ++t) {
        // spikes from step-entry membranes; in-place m <- lk*m - th*s (+C0 for l0)
        spike_pool_m0<<<4096, 256, 0, stream>>>(m0, C0, h0, th, lk);
        im2col16<<<1152, 128, 0, stream>>>(h0, A3, 64);
        spike_fc<<<8192, 256, 0, stream>>>(m3, s3, 2097152, th, lk, 1);
        im2col16<<<1152, 128, 0, stream>>>(s3, A6, 128);
        spike_pool_m6<<<2048, 256, 0, stream>>>(m6, h6, th, lk);
        spike_fc<<<1024, 256, 0, stream>>>(m9, s9, 262144, th, lk, 3);
        spike_fc<<<1024, 256, 0, stream>>>(m12, s12, 262144, th, lk, 4);

        // independent WMMA GEMMs accumulating into membranes
        gemm_nt_bf16_acc<<<dim3(256, 1), 128, 0, stream>>>(A3, w3r, m3, 576, 576, 128, 576);
        gemm_nt_bf16_acc<<<dim3(256, 1), 128, 0, stream>>>(A6, w6r, m6, 1152, 1152, 128, 1152);
        gemm_nt_bf16_acc<<<dim3(1, 32), 128, 0, stream>>>(h6, fc0r, m9, 8192, 8192, 4096, 8192);
        gemm_nt_bf16_acc<<<dim3(1, 32), 128, 0, stream>>>(s9, fc3r, m12, 4096, 4096, 4096, 4096);
        gemm_fc6_acc<<<dim3(1), 128, 0, stream>>>(s12, fc6r, mo, 4096);
    }
}
